// SwinBlock_13082470383781
// MI455X (gfx1250) — matrix operand order repaired and checked by reading, not compiled, not run on hardware
//
#include <hip/hip_runtime.h>
#include <cstddef>

// ---------------------------------------------------------------------------
// CDNA5 (gfx1250) wave32 WMMA types & helpers
// ---------------------------------------------------------------------------
typedef _Float16 half16v __attribute__((ext_vector_type(16)));
typedef _Float16 half8v  __attribute__((ext_vector_type(8)));
typedef float    float8v __attribute__((ext_vector_type(8)));

__device__ inline float8v wmma32(half16v a, half16v b, float8v c) {
#if defined(__HIP_DEVICE_COMPILE__)
  // D = A(16x32 f16) * B(32x16 f16) + C(16x16 f32)
  return __builtin_amdgcn_wmma_f32_16x16x32_f16(
      /*neg_a=*/false, a, /*neg_b=*/false, b,
      /*c_mod=*/(short)0, c, /*reuse_a=*/false, /*reuse_b=*/false);
#else
  (void)a; (void)b; return c;
#endif
}

// A-fragment (16x32 f16) from row-major f16 buffer, per ISA layout:
// lane m (0..15): M=m ; lanes 16..31: same M, upper K-halves.
// VGPR0..3 <- K = k0 + hi*8 + [0..7] ; VGPR4..7 <- K = k0 + 16 + hi*8 + [0..7]
__device__ inline half16v load_a(const _Float16* p, int stride, int row0,
                                 int k0, int lane) {
  const int m = lane & 15, hi = lane >> 4;
  const _Float16* q = p + (size_t)(row0 + m) * stride + k0 + hi * 8;
  half8v lo = *(const half8v*)q;          // 16B -> ds/global _load_b128
  half8v hh = *(const half8v*)(q + 16);   // 16B
  half16v a;
#pragma unroll
  for (int i = 0; i < 8; ++i) { a[i] = lo[i]; a[i + 8] = hh[i]; }
  return a;
}

// B-fragment (32x16 f16) from row-major [K][N] f16 buffer:
// lane l holds column N = n0 + l%16; K = k0 + hi*8 + [0..7], k0 + 16 + hi*8 + [0..7].
__device__ inline half16v load_b(const _Float16* p, int stride, int k0,
                                 int n0, int lane) {
  const _Float16* q = p + (size_t)(k0 + (lane >> 4) * 8) * stride + n0 + (lane & 15);
  half16v b;
#pragma unroll
  for (int i = 0; i < 8; ++i) { b[i] = q[(size_t)i * stride]; b[i + 8] = q[(size_t)(i + 16) * stride]; }
  return b;
}

// ---------------------------------------------------------------------------
// Kernel 0: convert weights f32 -> f16 into workspace (L2-resident afterwards)
// regions (halfs): qkv_w [0,27648) proj_w [27648,36864) mlp_w1 [36864,73728)
//                  mlp_w2 [73728,110592)
// ---------------------------------------------------------------------------
__global__ __launch_bounds__(256) void convert_weights_kernel(
    const float* __restrict__ qkvw, const float* __restrict__ projw,
    const float* __restrict__ w1, const float* __restrict__ w2,
    _Float16* __restrict__ dst) {
  int i = blockIdx.x * 256 + threadIdx.x;
  if (i < 27648)       dst[i] = (_Float16)qkvw[i];
  else if (i < 36864)  dst[i] = (_Float16)projw[i - 27648];
  else if (i < 73728)  dst[i] = (_Float16)w1[i - 36864];
  else if (i < 110592) dst[i] = (_Float16)w2[i - 73728];
}

// ---------------------------------------------------------------------------
// Kernel 1: fused LN1 + shifted-window attention + proj + residual*ls1
// One workgroup (128 thr = 4 wave32) per 8x8 window; 8*32*32 = 8192 WGs.
// ---------------------------------------------------------------------------
__global__ __launch_bounds__(128) void swin_attn_kernel(
    const float* __restrict__ x, const float* __restrict__ n1g,
    const float* __restrict__ n1b, const _Float16* __restrict__ qkvw,
    const float* __restrict__ qkvb, const _Float16* __restrict__ projw,
    const float* __restrict__ projb, const float* __restrict__ ls1,
    float* __restrict__ out) {
  // LDS regions with lifetime-based overlap (total 100352 B < 320 KB/WGP):
  //  hA [0,13312)        live ph1-2      (64x104 f16, LN'd tokens)
  //  qA [13312,26624)    live ph2-3      (64x104 f16, scaled q)
  //  kT [26624,38912)    live ph2-3      (3 x 32x64 f16, k transposed)
  //  vB [38912,51200)    live ph2-5      (64x96 f16)
  //  S  [51200,100352)   live ph3-4      (64x192 f32 logits)
  //  P  [13312,37888)    live ph4-5      (64x192 f16) -- reuses qA/kT
  //  oA [51200,64512)    live ph5-6      (64x104 f16) -- reuses S
  __shared__ __align__(32) char smem[100352];
  _Float16* hA = (_Float16*)(smem + 0);
  _Float16* qA = (_Float16*)(smem + 13312);
  _Float16* kT = (_Float16*)(smem + 26624);
  _Float16* vB = (_Float16*)(smem + 38912);
  float*    S  = (float*)(smem + 51200);
  _Float16* P  = (_Float16*)(smem + 13312);
  _Float16* oA = (_Float16*)(smem + 51200);

  const int tid  = threadIdx.x;
  const int lane = tid & 31;
  const int wv   = tid >> 5;  // wave id 0..3
  const int wg   = blockIdx.x;
  const int bb   = wg >> 10;        // batch
  const int wy   = (wg >> 5) & 31;  // window row
  const int wx   = wg & 31;         // window col

  // ---- Phase 1: gather shifted window + LayerNorm1 -> hA (f16) -------------
  if (tid < 64) {
    const int ty = tid >> 3, tx = tid & 7;
    const int gr = (wy * 8 + ty + 4) & 255;  // roll(-SHIFT)
    const int gc = (wx * 8 + tx + 4) & 255;
    const float* xp = x + (((size_t)bb * 256 + gr) * 256 + gc) * 96;
    float s = 0.f, s2 = 0.f;
#pragma unroll
    for (int i = 0; i < 24; ++i) {
      float4 f = ((const float4*)xp)[i];
      s  += f.x + f.y + f.z + f.w;
      s2 += f.x * f.x + f.y * f.y + f.z * f.z + f.w * f.w;
    }
    const float mean = s * (1.f / 96.f);
    const float var  = s2 * (1.f / 96.f) - mean * mean;
    const float rstd = rsqrtf(var + 1e-5f);
    for (int i = 0; i < 96; ++i) {
      float h = (xp[i] - mean) * rstd * n1g[i] + n1b[i];
      hA[tid * 104 + i] = (_Float16)h;
    }
  }
  __syncthreads();

  // ---- Phase 2: QKV GEMM (64x96)x(96x288), scatter q/kT/v ------------------
  {
    const int mt = wv;  // wave owns one 16-row M tile, reuses A across 18 N tiles
    const half16v a0 = load_a(hA, 104, mt * 16, 0, lane);
    const half16v a1 = load_a(hA, 104, mt * 16, 32, lane);
    const half16v a2 = load_a(hA, 104, mt * 16, 64, lane);
    const int n = lane & 15, hi = lane >> 4;
    for (int nt = 0; nt < 18; ++nt) {
      float8v acc = {};
      acc = wmma32(a0, load_b(qkvw, 288, 0,  nt * 16, lane), acc);
      acc = wmma32(a1, load_b(qkvw, 288, 32, nt * 16, lane), acc);
      acc = wmma32(a2, load_b(qkvw, 288, 64, nt * 16, lane), acc);
      const int col   = nt * 16 + n;     // 0..287
      const float bia = qkvb[col];
      const int which = col / 96;        // 0=q 1=k 2=v (tile-uniform: 96=6*16)
      const int cc    = col % 96;        // channel in [0,96)
      const int head  = cc >> 5, hc = cc & 31;
#pragma unroll
      for (int r = 0; r < 8; ++r) {
        const int tok = mt * 16 + hi * 8 + r;
        const float v = acc[r] + bia;
        if (which == 0)       qA[tok * 104 + cc] = (_Float16)(v * 0.1767766952966369f);
        else if (which == 1)  kT[head * 2048 + hc * 64 + tok] = (_Float16)v;
        else                  vB[tok * 96 + cc] = (_Float16)v;
      }
    }
  }
  __syncthreads();

  // ---- Phase 3: S = q @ k^T per head (3 heads x 4x4 tiles, K=32) -----------
  for (int j = wv; j < 48; j += 4) {
    const int head = j / 16, rem = j % 16, mt = rem >> 2, nt = rem & 3;
    half16v a = load_a(qA, 104, mt * 16, head * 32, lane);
    half16v b = load_b(kT + head * 2048, 64, 0, nt * 16, lane);
    float8v acc = {};
    acc = wmma32(a, b, acc);
    const int n = lane & 15, hi = lane >> 4;
#pragma unroll
    for (int r = 0; r < 8; ++r)
      S[(size_t)(mt * 16 + hi * 8 + r) * 192 + head * 64 + nt * 16 + n] = acc[r];
  }
  __syncthreads();

  // ---- Phase 4: row softmax over 64 cols; 192 (token,head) rows ------------
  for (int s = tid; s < 192; s += 128) {
    const int tok = s & 63, head = s >> 6;
    float* row = &S[(size_t)tok * 192 + head * 64];
    _Float16* prow = &P[(size_t)tok * 192 + head * 64];
    float mx = row[0];
    for (int j = 1; j < 64; ++j) mx = fmaxf(mx, row[j]);
    float sum = 0.f;
    for (int j = 0; j < 64; ++j) { float e = __expf(row[j] - mx); sum += e; row[j] = e; }
    const float inv = 1.f / sum;
    for (int j = 0; j < 64; ++j) prow[j] = (_Float16)(row[j] * inv);
  }
  __syncthreads();

  // ---- Phase 5: O = P @ V per head (3 x 4x2 tiles, K=64 = 2 WMMA) ----------
  for (int j = wv; j < 24; j += 4) {
    const int head = j / 8, rem = j % 8, mt = rem >> 1, nt = rem & 1;
    float8v acc = {};
    acc = wmma32(load_a(P, 192, mt * 16, head * 64, lane),
                 load_b(vB, 96, 0, head * 32 + nt * 16, lane), acc);
    acc = wmma32(load_a(P, 192, mt * 16, head * 64 + 32, lane),
                 load_b(vB, 96, 32, head * 32 + nt * 16, lane), acc);
    const int n = lane & 15, hi = lane >> 4;
#pragma unroll
    for (int r = 0; r < 8; ++r)
      oA[(mt * 16 + hi * 8 + r) * 104 + head * 32 + nt * 16 + n] = (_Float16)acc[r];
  }
  __syncthreads();

  // ---- Phase 6: proj (96x96) + bias, x1 = x + ls1*(.) -> global ------------
  for (int j = wv; j < 24; j += 4) {
    const int mt = j / 6, nt = j % 6;
    const half16v a0 = load_a(oA, 104, mt * 16, 0, lane);
    const half16v a1 = load_a(oA, 104, mt * 16, 32, lane);
    const half16v a2 = load_a(oA, 104, mt * 16, 64, lane);
    float8v acc = {};
    acc = wmma32(a0, load_b(projw, 96, 0,  nt * 16, lane), acc);
    acc = wmma32(a1, load_b(projw, 96, 32, nt * 16, lane), acc);
    acc = wmma32(a2, load_b(projw, 96, 64, nt * 16, lane), acc);
    const int n = lane & 15, hi = lane >> 4;
    const int col = nt * 16 + n;
    const float bia = projb[col];
    const float g   = ls1[col];
#pragma unroll
    for (int r = 0; r < 8; ++r) {
      const int tok = mt * 16 + hi * 8 + r;
      const int ty = tok >> 3, tx = tok & 7;
      const int gr = (wy * 8 + ty + 4) & 255;
      const int gc = (wx * 8 + tx + 4) & 255;
      const size_t gi = (((size_t)bb * 256 + gr) * 256 + gc) * 96 + col;
      out[gi] = x[gi] + g * (acc[r] + bia);
    }
  }
}

// ---------------------------------------------------------------------------
// Kernel 2: fused LN2 + MLP(96->384, exact GELU, 384->96) + residual*ls2.
// 64 consecutive tokens per WG, in-place on x1/out (each element owned by
// exactly one lane, so read-then-write is race-free).
// ---------------------------------------------------------------------------
__global__ __launch_bounds__(128) void swin_mlp_kernel(
    const float* x1, const float* __restrict__ n2g,
    const float* __restrict__ n2b, const _Float16* __restrict__ w1,
    const float* __restrict__ b1, const _Float16* __restrict__ w2,
    const float* __restrict__ b2, const float* __restrict__ ls2, float* out) {
  __shared__ __align__(32) _Float16 mA[64 * 104];  // LN2 tokens (f16)
  __shared__ __align__(32) _Float16 gA[64 * 392];  // GELU hidden (f16)

  const int tid = threadIdx.x;
  const int lane = tid & 31;
  const int wv = tid >> 5;
  const size_t base_tok = (size_t)blockIdx.x * 64;

  // ---- Phase 1: LayerNorm2 -> mA -------------------------------------------
  if (tid < 64) {
    const float* xp = x1 + (base_tok + tid) * 96;
    float s = 0.f, s2 = 0.f;
#pragma unroll
    for (int i = 0; i < 24; ++i) {
      float4 f = ((const float4*)xp)[i];
      s  += f.x + f.y + f.z + f.w;
      s2 += f.x * f.x + f.y * f.y + f.z * f.z + f.w * f.w;
    }
    const float mean = s * (1.f / 96.f);
    const float var  = s2 * (1.f / 96.f) - mean * mean;
    const float rstd = rsqrtf(var + 1e-5f);
    for (int i = 0; i < 96; ++i) {
      float h = (xp[i] - mean) * rstd * n2g[i] + n2b[i];
      mA[tid * 104 + i] = (_Float16)h;
    }
  }
  __syncthreads();

  // ---- Phase 2: GEMM1 (64x96)x(96x384) + exact GELU -> gA ------------------
  {
    const int mt = wv;
    const half16v a0 = load_a(mA, 104, mt * 16, 0, lane);
    const half16v a1 = load_a(mA, 104, mt * 16, 32, lane);
    const half16v a2 = load_a(mA, 104, mt * 16, 64, lane);
    const int n = lane & 15, hi = lane >> 4;
    for (int nt = 0; nt < 24; ++nt) {
      float8v acc = {};
      acc = wmma32(a0, load_b(w1, 384, 0,  nt * 16, lane), acc);
      acc = wmma32(a1, load_b(w1, 384, 32, nt * 16, lane), acc);
      acc = wmma32(a2, load_b(w1, 384, 64, nt * 16, lane), acc);
      const int col = nt * 16 + n;
      const float bia = b1[col];
#pragma unroll
      for (int r = 0; r < 8; ++r) {
        const float v = acc[r] + bia;
        const float ge = 0.5f * v * (1.f + erff(v * 0.7071067811865475f));
        gA[(mt * 16 + hi * 8 + r) * 392 + col] = (_Float16)ge;
      }
    }
  }
  __syncthreads();

  // ---- Phase 3: GEMM2 (64x384)x(384x96) + residual -------------------------
  for (int j = wv; j < 24; j += 4) {
    const int mt = j / 6, nt = j % 6;
    float8v acc = {};
#pragma unroll
    for (int kt = 0; kt < 12; ++kt) {
      acc = wmma32(load_a(gA, 392, mt * 16, kt * 32, lane),
                   load_b(w2, 96, kt * 32, nt * 16, lane), acc);
    }
    const int n = lane & 15, hi = lane >> 4;
    const int col = nt * 16 + n;
    const float bia = b2[col];
    const float g   = ls2[col];
#pragma unroll
    for (int r = 0; r < 8; ++r) {
      const int tok = mt * 16 + hi * 8 + r;
      const size_t gi = (base_tok + tok) * 96 + col;
      out[gi] = x1[gi] + g * (acc[r] + bia);
    }
  }
}

// ---------------------------------------------------------------------------
// Launch
// ---------------------------------------------------------------------------
extern "C" void kernel_launch(void* const* d_in, const int* in_sizes, int n_in,
                              void* d_out, int out_size, void* d_ws,
                              size_t ws_size, hipStream_t stream) {
  (void)in_sizes; (void)n_in; (void)out_size; (void)ws_size;
  const float* x      = (const float*)d_in[0];
  const float* n1g    = (const float*)d_in[1];
  const float* n1b    = (const float*)d_in[2];
  const float* qkv_w  = (const float*)d_in[3];
  const float* qkv_b  = (const float*)d_in[4];
  const float* proj_w = (const float*)d_in[5];
  const float* proj_b = (const float*)d_in[6];
  const float* ls1    = (const float*)d_in[7];
  const float* n2g    = (const float*)d_in[8];
  const float* n2b    = (const float*)d_in[9];
  const float* mlp_w1 = (const float*)d_in[10];
  const float* mlp_b1 = (const float*)d_in[11];
  const float* mlp_w2 = (const float*)d_in[12];
  const float* mlp_b2 = (const float*)d_in[13];
  const float* ls2    = (const float*)d_in[14];
  float* out = (float*)d_out;

  _Float16* wbuf    = (_Float16*)d_ws;
  _Float16* qkvw16  = wbuf;
  _Float16* projw16 = wbuf + 27648;
  _Float16* w1h     = wbuf + 36864;
  _Float16* w2h     = wbuf + 73728;

  convert_weights_kernel<<<432, 256, 0, stream>>>(qkv_w, proj_w, mlp_w1,
                                                  mlp_w2, wbuf);
  // 8 batches * 32*32 windows = 8192 WGs, 4 wave32 each
  swin_attn_kernel<<<8192, 128, 0, stream>>>(x, n1g, n1b, qkvw16, qkv_b,
                                             projw16, proj_b, ls1, out);
  // 524288 tokens / 64 per WG = 8192 WGs
  swin_mlp_kernel<<<8192, 128, 0, stream>>>(out, n2g, n2b, w1h, mlp_b1, w2h,
                                            mlp_b2, ls2, out);
}
